// AttentionLSTM_36893769073189
// MI455X (gfx1250) — matrix operand order repaired and checked by reading, not compiled, not run on hardware
//
#include <hip/hip_runtime.h>
#include <hip/hip_bf16.h>

// Problem sizes (fixed by reference)
#define NB    256            // batch
#define TT    128            // time steps
#define DD    512            // input dim
#define HH    1024           // hidden dim
#define FOURH 4096           // 4*H
#define KC    2560           // D + H + H  (fused K dimension)
#define KCH   64             // K-chunk staged in LDS per buffer

typedef __attribute__((ext_vector_type(16))) __bf16 v16bf;
typedef __attribute__((ext_vector_type(8)))  __bf16 v8bf;
typedef __attribute__((ext_vector_type(8)))  float  v8f;
typedef __attribute__((ext_vector_type(4)))  unsigned int u32x4;
typedef __attribute__((ext_vector_type(8)))  int i32x8;
typedef __attribute__((ext_vector_type(4)))  int i32x4;

union V16 { v16bf v; v8bf h[2]; };

// ---------------------------------------------------------------------------
// Tensor Data Mover: async 2D tile load Global -> LDS.
// D# built per CDNA5 ISA ch.8: group0 = {flags, lds_addr, global_addr, type=2},
// group1 = {data_size=2B, tensor_dim0/1, tile_dim0/1, dim0_stride}.
// Tracked by TENSORcnt; issuing wave waits with s_wait_tensorcnt.
// ---------------------------------------------------------------------------
__device__ __forceinline__ void tdm_load_2d(unsigned lds_off, const void* gptr,
                                            unsigned tile_k, unsigned tile_rows,
                                            unsigned tensor_d0, unsigned tensor_d1,
                                            unsigned stride_elems) {
  unsigned long long ga = (unsigned long long)gptr;
  u32x4 g0;
  g0.x = 0x1u;                                   // count=1 (valid user D#)
  g0.y = lds_off;                                // LDS byte address
  g0.z = (unsigned)(ga & 0xffffffffu);           // global_addr[31:0]
  g0.w = (unsigned)((ga >> 32) & 0x01ffffffu) | (2u << 30);  // addr[56:32], type=2
  i32x8 g1;
  g1[0] = (int)(1u << 16);                       // wg_mask=0, data_size=1 (2 bytes)
  g1[1] = (int)((tensor_d0 & 0xffffu) << 16);    // barrier_addr=0 | tensor_dim0 lo16
  g1[2] = (int)((tensor_d0 >> 16) | ((tensor_d1 & 0xffffu) << 16));
  g1[3] = (int)((tensor_d1 >> 16) | (tile_k << 16));   // tensor_dim1 hi16 | tile_dim0
  g1[4] = (int)(tile_rows & 0xffffu);            // tile_dim1 | tile_dim2=0
  g1[5] = (int)stride_elems;                     // tensor_dim0_stride lo32
  g1[6] = 0;                                     // stride hi16 | dim1_stride lo16
  g1[7] = 0;                                     // dim1_stride hi32
  i32x4 z4 = {0, 0, 0, 0};
#if defined(__clang_major__) && (__clang_major__ >= 23)
  i32x8 z8 = {0, 0, 0, 0, 0, 0, 0, 0};
  __builtin_amdgcn_tensor_load_to_lds(g0, g1, z4, z4, z8, 0);
#else
  __builtin_amdgcn_tensor_load_to_lds(g0, g1, z4, z4, 0);
#endif
}

__device__ __forceinline__ unsigned lds_offset(const void* p) {
  // generic LDS pointer: low 32 bits are the wave-relative LDS byte address
  return (unsigned)(unsigned long long)p;
}

// ---------------------------------------------------------------------------
// Pack [Wx; Wh; Wattn] (2560 x 4096, fp32 row-major) into bf16 TRANSPOSED
// layout WT[col][k] (4096 x 2560) so WMMA B-fragments load contiguously.
// ---------------------------------------------------------------------------
__global__ void k_build_wt(const float* __restrict__ Wx,
                           const float* __restrict__ Wh,
                           const float* __restrict__ Wa,
                           __bf16* __restrict__ WT) {
  int col = blockIdx.x;  // 0..4095
  __bf16* dst = WT + (size_t)col * KC;
  for (int k = threadIdx.x; k < KC; k += blockDim.x) {
    float v;
    if (k < DD)           v = Wx[(size_t)k * FOURH + col];
    else if (k < DD + HH) v = Wh[(size_t)(k - DD) * FOURH + col];
    else                  v = Wa[(size_t)(k - DD - HH) * FOURH + col];
    dst[k] = (__bf16)v;
  }
}

// ---------------------------------------------------------------------------
// x (N,T,D) fp32 -> bf16 (one time)
// ---------------------------------------------------------------------------
__global__ void k_xbf(const float* __restrict__ x, __bf16* __restrict__ xb,
                      int n) {
  int i = blockIdx.x * blockDim.x + threadIdx.x;
  if (i < n) xb[i] = (__bf16)x[i];
}

// ---------------------------------------------------------------------------
// h0 = c0 = mean over 16 spatial locations of A (N,H,16)
// ---------------------------------------------------------------------------
__global__ void k_init(const float* __restrict__ A, float* __restrict__ h,
                       float* __restrict__ c) {
  int n = blockIdx.x;
  for (int j = threadIdx.x; j < HH; j += blockDim.x) {
    const float* p = A + ((size_t)n * HH + j) * 16;
    float s = 0.f;
#pragma unroll
    for (int q = 0; q < 16; ++q) s += p[q];
    s *= (1.f / 16.f);
    h[(size_t)n * HH + j] = s;
    c[(size_t)n * HH + j] = s;
  }
}

// ---------------------------------------------------------------------------
// Per-step prepare: attention (scores -> softmax -> attn) and assemble the
// bf16 fused-A row  acat[n] = [ x_t(n) | h(n) | attn(n) ]   (N x 2560)
// One block per batch element n; 256 threads; LDS tree reduction.
// ---------------------------------------------------------------------------
__global__ __launch_bounds__(256)
void k_prepare(const float* __restrict__ A, const float* __restrict__ h,
               const __bf16* __restrict__ xb, __bf16* __restrict__ acat,
               int t) {
  __shared__ float red[256 * 16];
  __shared__ float w[16];
  const int n = blockIdx.x;
  const int tid = threadIdx.x;
  const float* Af = A + (size_t)n * HH * 16;
  const float* hn = h + (size_t)n * HH;

  float part[16];
#pragma unroll
  for (int s = 0; s < 16; ++s) part[s] = 0.f;
  float hv[4];
#pragma unroll
  for (int jj = 0; jj < 4; ++jj) {
    int j = tid + jj * 256;
    float hj = hn[j];
    hv[jj] = hj;
    const float* ar = Af + (size_t)j * 16;
#pragma unroll
    for (int s = 0; s < 16; ++s) part[s] += hj * ar[s];
  }
#pragma unroll
  for (int s = 0; s < 16; ++s) red[tid * 16 + s] = part[s];
  __syncthreads();
  for (int off = 128; off >= 1; off >>= 1) {
    if (tid < off) {
#pragma unroll
      for (int s = 0; s < 16; ++s)
        red[tid * 16 + s] += red[(tid + off) * 16 + s];
    }
    __syncthreads();
  }
  if (tid == 0) {
    const float scale = rsqrtf((float)HH);
    float sc[16], mx = -3.0e38f;
#pragma unroll
    for (int s = 0; s < 16; ++s) { sc[s] = red[s] * scale; mx = fmaxf(mx, sc[s]); }
    float sum = 0.f;
#pragma unroll
    for (int s = 0; s < 16; ++s) { sc[s] = __expf(sc[s] - mx); sum += sc[s]; }
    float inv = 1.f / sum;
#pragma unroll
    for (int s = 0; s < 16; ++s) w[s] = sc[s] * inv;
  }
  __syncthreads();

  __bf16* arow = acat + (size_t)n * KC;
  // x_t slice (already bf16)
  const __bf16* xrow = xb + ((size_t)n * TT + t) * DD;
  for (int d = tid; d < DD; d += 256) arow[d] = xrow[d];
  // h (bf16) and attn = A_flat @ w (bf16)
#pragma unroll
  for (int jj = 0; jj < 4; ++jj) {
    int j = tid + jj * 256;
    const float* ar = Af + (size_t)j * 16;
    float at = 0.f;
#pragma unroll
    for (int s = 0; s < 16; ++s) at += ar[s] * w[s];
    arow[DD + j]      = (__bf16)hv[jj];
    arow[DD + HH + j] = (__bf16)at;
  }
}

// ---------------------------------------------------------------------------
// Fused gate GEMM:  gates (256 x 4096, f32) = acat (256x2560 bf16)
//                    @ WT^T (2560x4096 bf16, stored transposed) + bias
//
// Block: 256 threads = 8 waves (2 in M x 4 in N), block tile 32 x 128.
// TDM double-buffers A (32xKCH) and B (128xKCH) tiles into LDS; wave 0
// issues the DMA for chunk k+1 while all waves run WMMA on chunk k from
// LDS, then waits s_wait_tensorcnt and the block barriers.
//
// A-fragment (16-bit 16x32, ISA 7.12.2): lane l row = l&15,
//   elems 0..7  -> K = kk + (l>=16?16:0) + 0..7   (contiguous ds_load_b128)
//   elems 8..15 -> K = kk + (l>=16?16:0) + 8..15
// B-fragment (32x16): lane l col = l&15, K = kk + (l>=16?16:0) + 0..15
// ---------------------------------------------------------------------------
__global__ __launch_bounds__(256)
void k_gates(const __bf16* __restrict__ Acat, const __bf16* __restrict__ WT,
             const float* __restrict__ bias, float* __restrict__ out) {
  __shared__ __align__(16) __bf16 sA[2][32 * KCH];    //  8 KB
  __shared__ __align__(16) __bf16 sB[2][128 * KCH];   // 32 KB

  const int lane = threadIdx.x & 31;
  const int wave = threadIdx.x >> 5;   // 0..7
  const int wM = wave >> 2;            // 0..1
  const int wN = wave & 3;             // 0..3
  const int rowBase = blockIdx.y * 32;
  const int colBase = blockIdx.x * 128;

  const int l15  = lane & 15;
  const int half = lane >> 4;

  v8f c0 = {};
  v8f c1 = {};

  // Preload chunk 0
  if (wave == 0) {
    tdm_load_2d(lds_offset(&sA[0][0]), Acat + (size_t)rowBase * KC,
                KCH, 32, KC, NB, KC);
    tdm_load_2d(lds_offset(&sB[0][0]), WT + (size_t)colBase * KC,
                KCH, 128, KC, FOURH, KC);
    __builtin_amdgcn_s_wait_tensorcnt(0);
  }
  __syncthreads();

  int buf = 0;
  for (int k0 = 0; k0 < KC; k0 += KCH) {
    const int nbuf = buf ^ 1;
    // Kick off DMA for the next chunk while this one is consumed
    if (wave == 0 && (k0 + KCH) < KC) {
      tdm_load_2d(lds_offset(&sA[nbuf][0]),
                  Acat + (size_t)rowBase * KC + (k0 + KCH), KCH, 32, KC, NB, KC);
      tdm_load_2d(lds_offset(&sB[nbuf][0]),
                  WT + (size_t)colBase * KC + (k0 + KCH), KCH, 128, KC, FOURH, KC);
    }

    const __bf16* a_base  = &sA[buf][(wM * 16 + l15) * KCH];
    const __bf16* b0_base = &sB[buf][(wN * 32 + l15) * KCH];
    const __bf16* b1_base = &sB[buf][(wN * 32 + 16 + l15) * KCH];
#pragma unroll
    for (int kk = 0; kk < KCH; kk += 32) {
      V16 a, b0f, b1f;
      const int ka = kk + half * 16;
      const int kb = kk + half * 16;
      a.h[0]   = *(const v8bf*)(a_base + ka);
      a.h[1]   = *(const v8bf*)(a_base + ka + 8);
      b0f.h[0] = *(const v8bf*)(b0_base + kb);
      b0f.h[1] = *(const v8bf*)(b0_base + kb + 8);
      b1f.h[0] = *(const v8bf*)(b1_base + kb);
      b1f.h[1] = *(const v8bf*)(b1_base + kb + 8);
      c0 = __builtin_amdgcn_wmma_f32_16x16x32_bf16(false, a.v, false, b0f.v,
                                                   (short)0, c0, false, false);
      c1 = __builtin_amdgcn_wmma_f32_16x16x32_bf16(false, a.v, false, b1f.v,
                                                   (short)0, c1, false, false);
    }

    if (wave == 0) __builtin_amdgcn_s_wait_tensorcnt(0);
    __syncthreads();
    buf = nbuf;
  }

  // C/D layout: VGPR r, lane l -> M = r + (l>=16 ? 8 : 0), N = l&15
  const int cn = l15;
  const int colW = colBase + wN * 32;
  const int rowW = rowBase + wM * 16;
  const float bb0 = bias[colW + cn];
  const float bb1 = bias[colW + 16 + cn];
#pragma unroll
  for (int r = 0; r < 8; ++r) {
    int m = rowW + half * 8 + r;
    out[(size_t)m * FOURH + colW + cn]      = c0[r] + bb0;
    out[(size_t)m * FOURH + colW + 16 + cn] = c1[r] + bb1;
  }
}

// ---------------------------------------------------------------------------
// Pointwise LSTM cell update; writes h into output (N,T,H)
// ---------------------------------------------------------------------------
__global__ __launch_bounds__(256)
void k_point(const float* __restrict__ gates, float* __restrict__ h,
             float* __restrict__ c, float* __restrict__ out, int t) {
  int i = blockIdx.x * blockDim.x + threadIdx.x;  // over N*H
  int n = i >> 10;
  int j = i & (HH - 1);
  const float* g = gates + (size_t)n * FOURH;
  float ai = g[j];
  float af = g[HH + j];
  float ao = g[2 * HH + j];
  float ag = g[3 * HH + j];
  float ig = 1.f / (1.f + __expf(-ai));
  float fg = 1.f / (1.f + __expf(-af));
  float og = 1.f / (1.f + __expf(-ao));
  float gg = tanhf(ag);
  float cc = fg * c[i] + ig * gg;
  float hh = og * tanhf(cc);
  c[i] = cc;
  h[i] = hh;
  out[((size_t)n * TT + t) * HH + j] = hh;
}

// ---------------------------------------------------------------------------
// Host launcher
// ---------------------------------------------------------------------------
extern "C" void kernel_launch(void* const* d_in, const int* in_sizes, int n_in,
                              void* d_out, int out_size, void* d_ws,
                              size_t ws_size, hipStream_t stream) {
  const float* x  = (const float*)d_in[0];   // (N,T,D)
  const float* A  = (const float*)d_in[1];   // (N,H,4,4)
  const float* Wx = (const float*)d_in[2];   // (D,4H)
  const float* Wh = (const float*)d_in[3];   // (H,4H)
  const float* Wa = (const float*)d_in[4];   // (H,4H)
  const float* b  = (const float*)d_in[5];   // (4H,)
  float* out = (float*)d_out;                // (N,T,H)

  char* ws = (char*)d_ws;
  size_t off = 0;
  auto alloc = [&](size_t bytes) -> void* {
    void* p = ws + off;
    off += (bytes + 255) & ~(size_t)255;
    return p;
  };
  __bf16* WT   = (__bf16*)alloc((size_t)FOURH * KC * 2);  // 20 MB
  __bf16* xb   = (__bf16*)alloc((size_t)NB * TT * DD * 2);// 32 MB
  __bf16* acat = (__bf16*)alloc((size_t)NB * KC * 2);     // 1.25 MB
  float* hbuf  = (float*)alloc((size_t)NB * HH * 4);      // 1 MB
  float* cbuf  = (float*)alloc((size_t)NB * HH * 4);      // 1 MB
  float* gates = (float*)alloc((size_t)NB * FOURH * 4);   // 4 MB

  // One-time packing / init
  k_build_wt<<<FOURH, 256, 0, stream>>>(Wx, Wh, Wa, WT);
  k_xbf<<<(NB * TT * DD) / 256, 256, 0, stream>>>(x, xb, NB * TT * DD);
  k_init<<<NB, 256, 0, stream>>>(A, hbuf, cbuf);

  // Sequential recurrence over T
  for (int t = 0; t < TT; ++t) {
    k_prepare<<<NB, 256, 0, stream>>>(A, hbuf, xb, acat, t);
    k_gates<<<dim3(FOURH / 128, NB / 32), 256, 0, stream>>>(acat, WT, b, gates);
    k_point<<<(NB * HH) / 256, 256, 0, stream>>>(gates, hbuf, cbuf, out, t);
  }
}
